// CompositionalKoopmanOperators_3006477107521
// MI455X (gfx1250) — matrix operand order repaired and checked by reading, not compiled, not run on hardware
//
#include <hip/hip_runtime.h>
#include <math.h>

// ---------------------------------------------------------------------------
// CDNA5 (gfx1250, wave32) WMMA implementation of the Koopman propagation net.
// All GEMMs run on v_wmma_f32_16x16x32_f16 with 2x4 (last predictor GEMM 2x2) register blocking per
// wave (8 wmma per 6 fragment loads). The 64MB rel_eff tensor is never
// materialized: one block per (b,i) computes it in registers and reduces
// over senders through LDS (ds_add_f32).
// ---------------------------------------------------------------------------

typedef _Float16 half_t;
typedef __attribute__((ext_vector_type(16))) _Float16 v16h;
typedef __attribute__((ext_vector_type(8)))  _Float16 v8h;
typedef __attribute__((ext_vector_type(8)))  float    v8f;

// Dimensions (fixed by the reference)
#define BB   16
#define NN   64
#define NF   256
#define GG   32
#define NROW (BB*NN)        // 1024 object rows
#define RROW (BB*NN*NN)     // 65536 relation rows

// ---------------------------------------------------------------------------
// Fragment loaders (wave32); contiguous k order, element s = K 16*kh + s, same in A and B.
// A: 16x32 f16, row-major source. lane: row = lane&15, khalf = lane>>4.
//    VGPR 0..3 hold K = 16*kh .. 16*kh+7 (16B), VGPR 4..7 hold +8 (16B).
__device__ __forceinline__ v16h load_frag_a(const half_t* __restrict__ tile,
                                            int lda, int lane) {
  const int row = lane & 15, kh = (lane >> 4) & 1;
  const half_t* p = tile + (size_t)row * lda + kh * 16;
  v8h lo = *(const v8h*)(p);
  v8h hi = *(const v8h*)(p + 8);
  v16h r;
#pragma unroll
  for (int i = 0; i < 8; ++i) { r[i] = lo[i]; r[i + 8] = hi[i]; }
  return r;
}

// B: 32x16 f16. Weights are stored TRANSPOSED ([N, K] row-major) so that the
// fragment (N = lane&15, K = 16*kh + 0..15) is one contiguous 32B load.
__device__ __forceinline__ v16h load_frag_b(const half_t* __restrict__ wt,
                                            int ldw, int lane) {
  const int n = lane & 15, kh = (lane >> 4) & 1;
  return *(const v16h*)(wt + (size_t)n * ldw + kh * 16);
}

// ---------------------------------------------------------------------------
// Prep kernels
// ---------------------------------------------------------------------------

// W [K,N] f32 -> WT [N,Kpad] f16 (zero-padded K)
__global__ void k_wtrans(const float* __restrict__ W, half_t* __restrict__ WT,
                         int K, int N, int Kpad) {
  int t = blockIdx.x * blockDim.x + threadIdx.x;
  if (t >= N * Kpad) return;
  int n = t / Kpad, k = t % Kpad;
  WT[t] = (k < K) ? (half_t)W[(size_t)k * N + n] : (half_t)0.f;
}

// obj_in [1024, 32] f16 : [attrs(4) | states(16) | pad(12)]
__global__ void k_obj_in(const float* __restrict__ attrs,
                         const float* __restrict__ states,
                         half_t* __restrict__ out) {
  int t = blockIdx.x * blockDim.x + threadIdx.x;
  if (t >= NROW * 32) return;
  int r = t >> 5, k = t & 31;
  float v = 0.f;
  if (k < 4)       v = attrs[r * 4 + k];
  else if (k < 20) v = states[r * 16 + (k - 4)];
  out[t] = (half_t)v;
}

// rel_in [65536, 64] f16 : [rel(9) | dstate(16) | recv_attr(4) | send_attr(4)]
__global__ void k_rel_in(const float* __restrict__ attrs,
                         const float* __restrict__ states,
                         const float* __restrict__ rel_attrs,
                         half_t* __restrict__ out) {
  int t = blockIdx.x * blockDim.x + threadIdx.x;
  if (t >= RROW * 64) return;
  int r = t >> 6, k = t & 63;
  int bi = r >> 6;                 // b*64 + i
  int j  = r & 63;
  int b  = r >> 12;
  int bj = (b << 6) | j;           // b*64 + j
  float v = 0.f;
  if (k < 9)       v = rel_attrs[(size_t)r * 9 + k];
  else if (k < 25) v = states[bi * 16 + (k - 9)] - states[bj * 16 + (k - 9)];
  else if (k < 29) v = attrs[bi * 4 + (k - 25)];
  else if (k < 33) v = attrs[bj * 4 + (k - 29)];
  out[t] = (half_t)v;
}

// ---------------------------------------------------------------------------
// Blocked WMMA GEMM: out = epilogue(A[M,K] @ WT[N,K]^T + bias)
// One wave per (16*MT x 16*NT) super-tile: per K-step, MT A-frags + NT B-frags
// feed MT*NT wmma ops. 8 waves per block; wave-uniform guard keeps EXEC
// all-ones around every v_wmma (ISA 7.12 requirement).
// ---------------------------------------------------------------------------
#define EPI_RELU_H 0   // relu, f16 output
#define EPI_TANH_F 1   // tanh, f32 output

template <int EPI, int MT, int NT>
__global__ void k_gemm(const half_t* __restrict__ A, int lda,
                       const half_t* __restrict__ WT, int ldw,
                       const float* __restrict__ bias,
                       void* __restrict__ outv, int ldo,
                       int superN, int totalSuper, int K) {
  const int wave = threadIdx.x >> 5, lane = threadIdx.x & 31;
  const int st = blockIdx.x * 8 + wave;
  if (st >= totalSuper) return;                // wave-uniform exit
  const int m0 = (st / superN) * (16 * MT);
  const int n0 = (st % superN) * (16 * NT);

  v8f acc[MT][NT] = {};
  for (int k0 = 0; k0 < K; k0 += 32) {
    v16h a[MT], b[NT];
#pragma unroll
    for (int mi = 0; mi < MT; ++mi)
      a[mi] = load_frag_a(A + (size_t)(m0 + 16 * mi) * lda + k0, lda, lane);
#pragma unroll
    for (int ni = 0; ni < NT; ++ni)
      b[ni] = load_frag_b(WT + (size_t)(n0 + 16 * ni) * ldw + k0, ldw, lane);
#pragma unroll
    for (int mi = 0; mi < MT; ++mi)
#pragma unroll
      for (int ni = 0; ni < NT; ++ni)
        acc[mi][ni] = __builtin_amdgcn_wmma_f32_16x16x32_f16(
            false, a[mi], false, b[ni], (short)0, acc[mi][ni], false, false);
  }

  const int kh = (lane >> 4) & 1;
#pragma unroll
  for (int ni = 0; ni < NT; ++ni) {
    const int n = n0 + 16 * ni + (lane & 15);
    const float bz = bias[n];
#pragma unroll
    for (int mi = 0; mi < MT; ++mi) {
#pragma unroll
      for (int rr = 0; rr < 8; ++rr) {
        const int m = m0 + 16 * mi + rr + kh * 8;   // C/D layout, ISA 7.12.2
        float v = acc[mi][ni][rr] + bz;
        if constexpr (EPI == EPI_RELU_H) {
          v = v > 0.f ? v : 0.f;
          ((half_t*)outv)[(size_t)m * ldo + n] = (half_t)v;
        } else {
          ((float*)outv)[(size_t)m * ldo + n] = tanhf(v);
        }
      }
    }
  }
}

// ---------------------------------------------------------------------------
// Fused relation-propagation + sender aggregation (K = 768 -> 256, summed).
// One block (8 waves) per receiver row r0 = b*64+i. Each wave owns a 32x64
// super-tile of the 64x256 rel_eff block (2 j-tiles x 4 n-tiles) computed in
// ONE K pass; tiles are relu'd, reduced over their sender rows in registers
// and ds_add_f32'd into LDS. rel_eff never touches memory.
// Output: objagg[r0] = [obj(256) | agg(256)] f16 for the pp GEMM.
// ---------------------------------------------------------------------------
__global__ void k_rp_agg(const half_t* __restrict__ rel_enc,  // [65536,256]
                         const half_t* __restrict__ obj,      // [1024,256]
                         const half_t* __restrict__ rpWT,     // [256,768]
                         const float*  __restrict__ rp_b,
                         half_t* __restrict__ objagg) {       // [1024,512]
  __shared__ float aggS[NF];
  __shared__ __align__(32) half_t recvS[NF];
  const int r0 = blockIdx.x;            // b*64 + i
  const int tid = threadIdx.x;
  aggS[tid]  = 0.f;
  recvS[tid] = obj[(size_t)r0 * NF + tid];   // receiver vector, block-shared
  __syncthreads();

  const int wave = tid >> 5, lane = tid & 31;
  const int bbase = r0 & ~63;           // b*64
  const int j0 = (wave & 1) * 32;       // 2 waves cover j = 0..63
  const int n0 = (wave >> 1) * 64;      // 4 waves cover n = 0..255

  v8f acc[2][4] = {};
  for (int kb = 0; kb < 24; ++kb) {
    const int k0 = kb * 32;
    v16h a[2], b[4];
#pragma unroll
    for (int mi = 0; mi < 2; ++mi) {
      const int j = j0 + 16 * mi;
      if (k0 < 256) {         // rel_enc features
        a[mi] = load_frag_a(rel_enc + (size_t)(r0 * 64 + j) * NF + k0, NF, lane);
      } else if (k0 < 512) {  // receiver obj (broadcast row, from LDS)
        a[mi] = load_frag_a(recvS + (k0 - 256), 0, lane);
      } else {                // sender obj
        a[mi] = load_frag_a(obj + (size_t)(bbase + j) * NF + (k0 - 512), NF, lane);
      }
    }
#pragma unroll
    for (int ni = 0; ni < 4; ++ni)
      b[ni] = load_frag_b(rpWT + (size_t)(n0 + 16 * ni) * 768 + k0, 768, lane);
#pragma unroll
    for (int mi = 0; mi < 2; ++mi)
#pragma unroll
      for (int ni = 0; ni < 4; ++ni)
        acc[mi][ni] = __builtin_amdgcn_wmma_f32_16x16x32_f16(
            false, a[mi], false, b[ni], (short)0, acc[mi][ni], false, false);
  }

#pragma unroll
  for (int ni = 0; ni < 4; ++ni) {
    const int n = n0 + 16 * ni + (lane & 15);
    const float bz = rp_b[n];
    float s = 0.f;
#pragma unroll
    for (int mi = 0; mi < 2; ++mi)
#pragma unroll
      for (int rr = 0; rr < 8; ++rr) {
        float v = acc[mi][ni][rr] + bz;
        s += v > 0.f ? v : 0.f;
      }
    s += __shfl_xor(s, 16, 32);         // combine M-halves (lane ^ 16)
    if (lane < 16) atomicAdd(&aggS[n], s);   // ds_add_f32
  }
  __syncthreads();
  objagg[(size_t)r0 * 512 + tid]       = recvS[tid];
  objagg[(size_t)r0 * 512 + 256 + tid] = (half_t)aggS[tid];
}

// ---------------------------------------------------------------------------
// Host launch
// ---------------------------------------------------------------------------
extern "C" void kernel_launch(void* const* d_in, const int* in_sizes, int n_in,
                              void* d_out, int out_size, void* d_ws, size_t ws_size,
                              hipStream_t stream) {
  const float* attrs     = (const float*)d_in[0];
  const float* states    = (const float*)d_in[1];
  const float* rel_attrs = (const float*)d_in[2];
  // d_in[3] = pstep (fixed at 2 by setup; host read would break graph capture)
  const float* enc_w0 = (const float*)d_in[4];
  const float* enc_b0 = (const float*)d_in[5];
  const float* enc_w1 = (const float*)d_in[6];
  const float* enc_b1 = (const float*)d_in[7];
  const float* rel_w0 = (const float*)d_in[8];
  const float* rel_b0 = (const float*)d_in[9];
  const float* rel_w1 = (const float*)d_in[10];
  const float* rel_b1 = (const float*)d_in[11];
  const float* rp_w   = (const float*)d_in[12];
  const float* rp_b   = (const float*)d_in[13];
  const float* pp_w   = (const float*)d_in[14];
  const float* pp_b   = (const float*)d_in[15];
  const float* pred_w0 = (const float*)d_in[16];
  const float* pred_b0 = (const float*)d_in[17];
  const float* pred_w1 = (const float*)d_in[18];
  const float* pred_b1 = (const float*)d_in[19];
  float* out = (float*)d_out;

  // Workspace bump allocator (total ~76 MB; fits MI455X 192MB L2)
  size_t off = 0;
  auto alloc = [&](size_t bytes) -> void* {
    void* p = (char*)d_ws + off;
    off = (off + bytes + 255) & ~(size_t)255;
    return p;
  };
  half_t* wtE0 = (half_t*)alloc(256 * 32  * 2);
  half_t* wtE1 = (half_t*)alloc(256 * 256 * 2);
  half_t* wtR0 = (half_t*)alloc(256 * 64  * 2);
  half_t* wtR1 = (half_t*)alloc(256 * 256 * 2);
  half_t* wtRP = (half_t*)alloc(256 * 768 * 2);
  half_t* wtPP = (half_t*)alloc(256 * 512 * 2);
  half_t* wtP0 = (half_t*)alloc(256 * 256 * 2);
  half_t* wtP1 = (half_t*)alloc(32  * 256 * 2);
  half_t* obj_in = (half_t*)alloc((size_t)NROW * 32 * 2);
  half_t* objA   = (half_t*)alloc((size_t)NROW * NF * 2);   // obj activations
  half_t* objB   = (half_t*)alloc((size_t)NROW * NF * 2);   // hidden buffer
  half_t* objagg = (half_t*)alloc((size_t)NROW * 512 * 2);  // [obj|agg]
  half_t* rel_in  = (half_t*)alloc((size_t)RROW * 64 * 2);
  half_t* relH    = (half_t*)alloc((size_t)RROW * NF * 2);
  half_t* rel_enc = (half_t*)alloc((size_t)RROW * NF * 2);

  auto cdiv = [](int a, int b) { return (a + b - 1) / b; };

  // 1) Transpose + f16-convert all weights (WT layout [N, Kpad])
  k_wtrans<<<cdiv(256 * 32, 256), 256, 0, stream>>>(enc_w0, wtE0, 20, 256, 32);
  k_wtrans<<<cdiv(256 * 256, 256), 256, 0, stream>>>(enc_w1, wtE1, 256, 256, 256);
  k_wtrans<<<cdiv(256 * 64, 256), 256, 0, stream>>>(rel_w0, wtR0, 33, 256, 64);
  k_wtrans<<<cdiv(256 * 256, 256), 256, 0, stream>>>(rel_w1, wtR1, 256, 256, 256);
  k_wtrans<<<cdiv(256 * 768, 256), 256, 0, stream>>>(rp_w, wtRP, 768, 256, 768);
  k_wtrans<<<cdiv(256 * 512, 256), 256, 0, stream>>>(pp_w, wtPP, 512, 256, 512);
  k_wtrans<<<cdiv(256 * 256, 256), 256, 0, stream>>>(pred_w0, wtP0, 256, 256, 256);
  k_wtrans<<<cdiv(32 * 256, 256), 256, 0, stream>>>(pred_w1, wtP1, 256, 32, 256);

  // 2) Build padded f16 feature rows
  k_obj_in<<<cdiv(NROW * 32, 256), 256, 0, stream>>>(attrs, states, obj_in);
  k_rel_in<<<cdiv(RROW * 64, 256), 256, 0, stream>>>(attrs, states, rel_attrs, rel_in);

  // 3) Object encoder: relu(relu(obj_in @ E0) @ E1)   (M=1024, 2x4 blocking)
  k_gemm<EPI_RELU_H, 2, 4><<<cdiv(32 * 4, 8), 256, 0, stream>>>(
      obj_in, 32, wtE0, 32, enc_b0, objB, NF, 4, 32 * 4, 32);
  k_gemm<EPI_RELU_H, 2, 4><<<cdiv(32 * 4, 8), 256, 0, stream>>>(
      objB, NF, wtE1, NF, enc_b1, objA, NF, 4, 32 * 4, 256);

  // 4) Relation encoder: relu(relu(rel_in @ R0) @ R1)  (M=65536)
  k_gemm<EPI_RELU_H, 2, 4><<<cdiv(2048 * 4, 8), 256, 0, stream>>>(
      rel_in, 64, wtR0, 64, rel_b0, relH, NF, 4, 2048 * 4, 64);
  k_gemm<EPI_RELU_H, 2, 4><<<cdiv(2048 * 4, 8), 256, 0, stream>>>(
      relH, NF, wtR1, NF, rel_b1, rel_enc, NF, 4, 2048 * 4, 256);

  // 5) Propagation, pstep = 2 (fused rp GEMM + sender aggregation, then pp)
  for (int step = 0; step < 2; ++step) {
    k_rp_agg<<<NROW, 256, 0, stream>>>(rel_enc, objA, wtRP, rp_b, objagg);
    k_gemm<EPI_RELU_H, 2, 4><<<cdiv(32 * 4, 8), 256, 0, stream>>>(
        objagg, 512, wtPP, 512, pp_b, objA, NF, 4, 32 * 4, 512);
  }

  // 6) Predictor: tanh(relu(obj @ P0) @ P1) -> d_out [16,64,32] f32
  k_gemm<EPI_RELU_H, 2, 4><<<cdiv(32 * 4, 8), 256, 0, stream>>>(
      objA, NF, wtP0, NF, pred_b0, objB, NF, 4, 32 * 4, 256);
  k_gemm<EPI_TANH_F, 2, 2><<<cdiv(32 * 1, 8), 256, 0, stream>>>(
      objB, NF, wtP1, NF, pred_b1, out, GG, 1, 32 * 1, 256);

  (void)in_sizes; (void)n_in; (void)out_size; (void)ws_size;
}
